// CausalSelfAttention_2731599200718
// MI455X (gfx1250) — matrix operand order repaired and checked by reading, not compiled, not run on hardware
//
#include <hip/hip_runtime.h>

#define TT 4096
#define CC 768
#define HD 64
#define NH 12
#define C3 2304

typedef __attribute__((ext_vector_type(16))) __bf16 bf16x16;
typedef __attribute__((ext_vector_type(8)))  float  f32x8;

union FragBF { bf16x16 v; unsigned int u[8]; };

static __device__ __forceinline__ unsigned short f2bf(float f) {
  unsigned int u = __float_as_uint(f);
  u += 0x7FFFu + ((u >> 16) & 1u);   // round-to-nearest-even
  return (unsigned short)(u >> 16);
}

// A-operand (16x32 bf16, M=lane%16): lane needs K pairs at
// k0 + 16*half + {0..15}. Row data contiguous in K (row-major, ld=K).
static __device__ __forceinline__ bf16x16
load_frag_a(const unsigned short* base, int row, int ld, int k0, int half) {
  const unsigned int* p = (const unsigned int*)(base + (size_t)row * ld + k0 + 16 * half);
  FragBF f;
#pragma unroll
  for (int vg = 0; vg < 8; ++vg) f.u[vg] = p[vg];
  return f.v;
}

// B-operand (32x16 bf16, N=lane%16): lane needs K pairs at k0 + 16*half + {0..15}
// => 8 contiguous dwords of "row n of B^T" (stored N-major, ld = K-extent).
static __device__ __forceinline__ bf16x16
load_frag_b(const unsigned short* base, int row, int ld, int k0, int half) {
  const unsigned int* p = (const unsigned int*)(base + (size_t)row * ld + k0 + 16 * half);
  FragBF f;
#pragma unroll
  for (int vg = 0; vg < 8; ++vg) f.u[vg] = p[vg];
  return f.v;
}

static __device__ __forceinline__ f32x8 wmma_bf16(bf16x16 a, bf16x16 b, f32x8 c) {
  return __builtin_amdgcn_wmma_f32_16x16x32_bf16(false, a, false, b, (short)0, c,
                                                 false, false);
}

// ---------------- conversion / transpose kernels ----------------

__global__ void cvt_bf16_kernel(const float* __restrict__ in,
                                unsigned short* __restrict__ out, int n) {
  int i = blockIdx.x * blockDim.x + threadIdx.x;
  if (i < n) out[i] = f2bf(in[i]);
}

// w is K x N row-major fp32; produce N x K row-major bf16 (B^T form)
__global__ void transpose_bf16_kernel(const float* __restrict__ w,
                                      unsigned short* __restrict__ wt, int K, int N) {
  int i = blockIdx.x * blockDim.x + threadIdx.x;
  if (i < K * N) {
    int nI = i / K, kI = i % K;
    wt[i] = f2bf(w[(size_t)kI * N + nI]);
  }
}

// ---------------- QKV GEMM + bias + RoPE ----------------
// wave = (32-token x 64-channel) tile: acc[2][4], every B-fragment feeds 2 WMMAs.
// grid: 128 supertiles x 36 chunks.

__global__ __launch_bounds__(256) void qkv_rope_kernel(
    const unsigned short* __restrict__ xb,   // 4096x768 bf16
    const unsigned short* __restrict__ wT,   // 2304x768 bf16 (w_attn^T)
    const float* __restrict__ bias,          // 2304 fp32
    unsigned short* __restrict__ qb,         // (H,T,hd) bf16, RoPE'd, *1/8
    unsigned short* __restrict__ kb,         // (H,T,hd) bf16, RoPE'd
    unsigned short* __restrict__ vb) {       // (H,hd,T) bf16 (transposed)
  const int lane = threadIdx.x & 31, wid = threadIdx.x >> 5;
  const int wave = blockIdx.x * 8 + wid;
  const int mt = wave / 36, chunk = wave % 36;
  const int nloc = lane & 15, half = lane >> 4;
  const int rbase = mt * 32;

  f32x8 acc[2][4] = {};
  for (int k0 = 0; k0 < CC; k0 += 32) {
    if (k0 + 32 < CC) {  // stream-ahead hints -> global_prefetch_b8
      __builtin_prefetch(xb + (size_t)(rbase + nloc) * CC + k0 + 32, 0, 3);
      __builtin_prefetch(xb + (size_t)(rbase + 16 + nloc) * CC + k0 + 32, 0, 3);
      __builtin_prefetch(wT + (size_t)(chunk * 64 + nloc) * CC + k0 + 32, 0, 3);
      __builtin_prefetch(wT + (size_t)(chunk * 64 + 32 + nloc) * CC + k0 + 32, 0, 3);
    }
    bf16x16 a0 = load_frag_a(xb, rbase + nloc, CC, k0, half);
    bf16x16 a1 = load_frag_a(xb, rbase + 16 + nloc, CC, k0, half);
#pragma unroll
    for (int j = 0; j < 4; ++j) {
      bf16x16 b = load_frag_b(wT, chunk * 64 + j * 16 + nloc, CC, k0, half);
      acc[0][j] = wmma_bf16(a0, b, acc[0][j]);
      acc[1][j] = wmma_bf16(a1, b, acc[1][j]);
    }
  }

  const int region = chunk / 12;   // 0=q, 1=k, 2=v
  const int h = chunk % 12;
  float bj[4];
#pragma unroll
  for (int j = 0; j < 4; ++j) bj[j] = bias[chunk * 64 + j * 16 + nloc];

#pragma unroll
  for (int mi = 0; mi < 2; ++mi) {
#pragma unroll
    for (int r = 0; r < 8; ++r) {
      const int t = rbase + mi * 16 + r + 8 * half;
      float vals[4];
#pragma unroll
      for (int j = 0; j < 4; ++j) vals[j] = acc[mi][j][r] + bj[j];
      if (region == 2) {
#pragma unroll
        for (int j = 0; j < 4; ++j)
          vb[h * (HD * TT) + (j * 16 + nloc) * TT + t] = f2bf(vals[j]);
      } else {
        unsigned short* dst = (region == 0) ? qb : kb;
        const float scl = (region == 0) ? 0.125f : 1.0f;  // fold 1/sqrt(hd) into q
#pragma unroll
        for (int j = 0; j < 4; ++j) {
          const int d = j * 16 + nloc;
          const int f = d & 31;
          // inv_freq = 10000^(-f/32) ; ln(10000)/32 = 0.287823136...
          const float ang = (float)t * __expf(-(float)f * 0.28782313662425572f);
          float sn, cs;
          __sincosf(ang, &sn, &cs);
          const float pv = vals[j ^ 2];          // paired channel d +/- 32
          const float o = vals[j] * cs + (d < 32 ? -pv : pv) * sn;
          dst[h * (TT * HD) + t * HD + d] = f2bf(o * scl);
        }
      }
    }
  }
}

// ---------------- flash attention ----------------
// wave = 16 query rows of one head; streams keys 32 at a time.

__global__ __launch_bounds__(256) void attn_kernel(
    const unsigned short* __restrict__ qb, const unsigned short* __restrict__ kb,
    const unsigned short* __restrict__ vb, unsigned short* __restrict__ yb) {
  __shared__ unsigned short pbuf[8 * 16 * 32];  // per-wave 16x32 bf16 P tile
  const int lane = threadIdx.x & 31, wid = threadIdx.x >> 5;
  const int wave = blockIdx.x * 8 + wid;
  const int h = wave >> 8, qtile = wave & 255;
  const int nloc = lane & 15, half = lane >> 4;
  const unsigned short* qh = qb + h * (TT * HD);
  const unsigned short* kh = kb + h * (TT * HD);
  const unsigned short* vh = vb + h * (HD * TT);
  unsigned short* pw = pbuf + wid * (16 * 32);

  const bf16x16 qA0 = load_frag_a(qh, qtile * 16 + nloc, HD, 0, half);
  const bf16x16 qA1 = load_frag_a(qh, qtile * 16 + nloc, HD, 32, half);

  f32x8 o[4] = {};
  float mr[8], lr[8];
#pragma unroll
  for (int r = 0; r < 8; ++r) { mr[r] = -1e30f; lr[r] = 0.0f; }
  const int tmax = qtile * 16 + 15;

  for (int sb = 0; sb <= tmax; sb += 32) {
    if (sb + 32 <= tmax) {  // prefetch next key/value chunk
      __builtin_prefetch(kh + (size_t)(sb + 32 + nloc) * HD, 0, 3);
      __builtin_prefetch(kh + (size_t)(sb + 48 + nloc) * HD, 0, 3);
      __builtin_prefetch(vh + (size_t)(nloc * 4) * TT + sb + 32, 0, 3);
    }
    f32x8 sc[2] = {};
#pragma unroll
    for (int c = 0; c < 2; ++c) {
      int krow = sb + c * 16 + nloc;
      krow = krow < TT ? krow : TT - 1;  // clamp; masked below anyway
      bf16x16 b0 = load_frag_b(kh, krow, HD, 0, half);
      bf16x16 b1 = load_frag_b(kh, krow, HD, 32, half);
      sc[c] = wmma_bf16(qA0, b0, sc[c]);
      sc[c] = wmma_bf16(qA1, b1, sc[c]);
    }
    // online softmax over C-layout rows (row M = r + 8*half, cols across 16 lanes)
#pragma unroll
    for (int r = 0; r < 8; ++r) {
      const int t = qtile * 16 + r + 8 * half;
      float x0 = (sb + nloc) <= t ? sc[0][r] : -1e30f;
      float x1 = (sb + 16 + nloc) <= t ? sc[1][r] : -1e30f;
      float a = fmaxf(x0, x1);
      a = fmaxf(a, __shfl_xor(a, 1, 32));
      a = fmaxf(a, __shfl_xor(a, 2, 32));
      a = fmaxf(a, __shfl_xor(a, 4, 32));
      a = fmaxf(a, __shfl_xor(a, 8, 32));
      const float mnew = fmaxf(mr[r], a);
      const float corr = __expf(mr[r] - mnew);
      const float p0 = __expf(x0 - mnew), p1 = __expf(x1 - mnew);
      pw[(r + 8 * half) * 32 + nloc] = f2bf(p0);
      pw[(r + 8 * half) * 32 + 16 + nloc] = f2bf(p1);
      float rs = p0 + p1;
      rs += __shfl_xor(rs, 1, 32);
      rs += __shfl_xor(rs, 2, 32);
      rs += __shfl_xor(rs, 4, 32);
      rs += __shfl_xor(rs, 8, 32);
      lr[r] = lr[r] * corr + rs;
      mr[r] = mnew;
#pragma unroll
      for (int j = 0; j < 4; ++j) o[j][r] *= corr;
    }
    __builtin_amdgcn_wave_barrier();       // LDS ops are in-order per wave
    const bf16x16 pA = load_frag_a(pw, nloc, 32, 0, half);
    __builtin_amdgcn_wave_barrier();
#pragma unroll
    for (int j = 0; j < 4; ++j) {
      bf16x16 vB = load_frag_b(vh, j * 16 + nloc, TT, sb, half);
      o[j] = wmma_bf16(pA, vB, o[j]);
    }
  }
#pragma unroll
  for (int r = 0; r < 8; ++r) {
    const int t = qtile * 16 + r + 8 * half;
    const float inv = 1.0f / lr[r];
#pragma unroll
    for (int j = 0; j < 4; ++j)
      yb[t * CC + h * HD + j * 16 + nloc] = f2bf(o[j][r] * inv);
  }
}

// ---------------- output projection ----------------
// wave = (32-token x 64-channel) tile; grid: 128 supertiles x 12 chunks.

__global__ __launch_bounds__(256) void proj_kernel(
    const unsigned short* __restrict__ yb,   // 4096x768 bf16
    const unsigned short* __restrict__ wT,   // 768x768 bf16 (w_proj^T)
    const float* __restrict__ bias, float* __restrict__ out) {
  const int lane = threadIdx.x & 31, wid = threadIdx.x >> 5;
  const int wave = blockIdx.x * 8 + wid;
  const int mt = wave / 12, chunk = wave % 12;
  const int nloc = lane & 15, half = lane >> 4;
  const int rbase = mt * 32;

  f32x8 acc[2][4] = {};
  for (int k0 = 0; k0 < CC; k0 += 32) {
    if (k0 + 32 < CC) {
      __builtin_prefetch(yb + (size_t)(rbase + nloc) * CC + k0 + 32, 0, 3);
      __builtin_prefetch(yb + (size_t)(rbase + 16 + nloc) * CC + k0 + 32, 0, 3);
      __builtin_prefetch(wT + (size_t)(chunk * 64 + nloc) * CC + k0 + 32, 0, 3);
      __builtin_prefetch(wT + (size_t)(chunk * 64 + 32 + nloc) * CC + k0 + 32, 0, 3);
    }
    bf16x16 a0 = load_frag_a(yb, rbase + nloc, CC, k0, half);
    bf16x16 a1 = load_frag_a(yb, rbase + 16 + nloc, CC, k0, half);
#pragma unroll
    for (int j = 0; j < 4; ++j) {
      bf16x16 b = load_frag_b(wT, chunk * 64 + j * 16 + nloc, CC, k0, half);
      acc[0][j] = wmma_bf16(a0, b, acc[0][j]);
      acc[1][j] = wmma_bf16(a1, b, acc[1][j]);
    }
  }
  float bj[4];
#pragma unroll
  for (int j = 0; j < 4; ++j) bj[j] = bias[chunk * 64 + j * 16 + nloc];
#pragma unroll
  for (int mi = 0; mi < 2; ++mi) {
#pragma unroll
    for (int r = 0; r < 8; ++r) {
      const int t = rbase + mi * 16 + r + 8 * half;
#pragma unroll
      for (int j = 0; j < 4; ++j)
        out[(size_t)t * CC + chunk * 64 + j * 16 + nloc] = acc[mi][j][r] + bj[j];
    }
  }
}

// ---------------- launcher ----------------

extern "C" void kernel_launch(void* const* d_in, const int* in_sizes, int n_in,
                              void* d_out, int out_size, void* d_ws, size_t ws_size,
                              hipStream_t stream) {
  (void)in_sizes; (void)n_in; (void)out_size; (void)ws_size;
  const float* x      = (const float*)d_in[0];
  const float* w_attn = (const float*)d_in[1];
  const float* b_attn = (const float*)d_in[2];
  const float* w_proj = (const float*)d_in[3];
  const float* b_proj = (const float*)d_in[4];

  unsigned short* p = (unsigned short*)d_ws;
  unsigned short* xb     = p; p += TT * CC;        // x in bf16
  unsigned short* wattnT = p; p += C3 * CC;        // w_attn^T bf16
  unsigned short* wprojT = p; p += CC * CC;        // w_proj^T bf16
  unsigned short* qb     = p; p += NH * TT * HD;   // q (H,T,hd)
  unsigned short* kb     = p; p += NH * TT * HD;   // k (H,T,hd)
  unsigned short* vb     = p; p += NH * TT * HD + 64;  // v (H,hd,T) + OOB pad
  unsigned short* yb     = p;                      // attn out bf16 (T,C)

  cvt_bf16_kernel<<<(TT * CC + 255) / 256, 256, 0, stream>>>(x, xb, TT * CC);
  transpose_bf16_kernel<<<(CC * C3 + 255) / 256, 256, 0, stream>>>(w_attn, wattnT, CC, C3);
  transpose_bf16_kernel<<<(CC * CC + 255) / 256, 256, 0, stream>>>(w_proj, wprojT, CC, CC);

  qkv_rope_kernel<<<(128 * 36) / 8, 256, 0, stream>>>(xb, wattnT, b_attn, qb, kb, vb);
  attn_kernel<<<(NH * 256) / 8, 256, 0, stream>>>(qb, kb, vb, yb);
  proj_kernel<<<(128 * 12) / 8, 256, 0, stream>>>(yb, wprojT, b_proj, (float*)d_out);
}
